// DartsCell_79328045957239
// MI455X (gfx1250) — matrix operand order repaired and checked by reading, not compiled, not run on hardware
//
#include <hip/hip_runtime.h>
#include <hip/hip_bf16.h>
#include <stdint.h>

// ---------------------------------------------------------------------------
// DARTS GNN cell for MI455X (gfx1250).
// Dense [n,64]x[64,64] GEMMs -> v_wmma_f32_16x16x32_bf16 (f32 accumulate),
// LDS tiles pre-swizzled into fragment order so reads are ds_load_b128.
// Edge gather/scatter (the HBM-bound critical path) -> fused atomic kernels.
// ---------------------------------------------------------------------------

typedef __attribute__((ext_vector_type(16))) __bf16 bf16x16;
typedef __attribute__((ext_vector_type(8)))  __bf16 bf16x8;
typedef __attribute__((ext_vector_type(2)))  __bf16 bf16x2;
typedef __attribute__((ext_vector_type(8)))  float  f32x8;

#define FEAT 64
#define NEG_SLOPE 0.2f

__device__ __forceinline__ unsigned encf(float f) {
  unsigned u = __float_as_uint(f);
  return (u & 0x80000000u) ? ~u : (u | 0x80000000u);   // monotone float->uint
}
__device__ __forceinline__ float decf(unsigned u) {
  unsigned b = (u & 0x80000000u) ? (u & 0x7fffffffu) : ~u;
  return __uint_as_float(b);
}
__device__ __forceinline__ float leaky(float x) { return x > 0.f ? x : NEG_SLOPE * x; }

// ---------------------------------------------------------------------------
// LDS tile writers (fragment-order swizzle, vectorized stores).
//
// sHf layout: [row:64][kk32:2][half:2][e:16]  (A fragments, contiguous per lane)
//   A 16-bit fragment element e -> H column kk32*32 + rr*16 + half*8 + (e&7),
//   where rr = e>>3.  So each (row,kk32,half) is two contiguous 8-col runs.
// sWf layout: [kk32:2][ct:4][m:16][half:2][e:16] (B fragments)
//   element e -> W[kk32*32 + (e>>3)*16 + half*8 + (e&7)][ct*16 + m].
// ---------------------------------------------------------------------------
__device__ __forceinline__ void stage_H(const float* __restrict__ H, __bf16* sHf,
                                        int rowBase, int n, int tid) {
  for (int r8 = tid; r8 < 512; r8 += 128) {
    int i    = r8 << 3;                 // linear bf16 index, multiple of 8
    int rr   = (i >> 3) & 1;
    int half = (i >> 4) & 1;
    int kk32 = (i >> 5) & 1;
    int row  = i >> 6;
    int colBase = kk32 * 32 + rr * 16 + half * 8;
    int r = rowBase + row;
    bf16x8 v;
    if (r < n) {
      const float4* p = (const float4*)(H + (size_t)r * FEAT + colBase);
      float4 a = p[0], b = p[1];
      v[0] = (__bf16)a.x; v[1] = (__bf16)a.y; v[2] = (__bf16)a.z; v[3] = (__bf16)a.w;
      v[4] = (__bf16)b.x; v[5] = (__bf16)b.y; v[6] = (__bf16)b.z; v[7] = (__bf16)b.w;
    } else {
      v = (bf16x8){};
    }
    *(bf16x8*)&sHf[i] = v;
  }
}

__device__ __forceinline__ void stage_W(const float* __restrict__ W, __bf16* sWf,
                                        int tid) {
  for (int t = tid; t < 2048; t += 128) {
    int i    = t << 1;                  // linear bf16 index, even
    int e    = i & 15;
    int half = (i >> 4) & 1;
    int m    = (i >> 5) & 15;
    int ct   = (i >> 9) & 3;
    int kk32 = i >> 11;
    int krow = kk32 * 32 + (e >> 3) * 16 + half * 8 + (e & 7);
    int col  = ct * 16 + m;
    bf16x2 v;
    v[0] = (__bf16)W[(size_t)krow * FEAT + col];
    v[1] = (__bf16)W[(size_t)(krow + 1) * FEAT + col];
    *(bf16x2*)&sWf[i] = v;
  }
}

// One wave computes a 16x64 slab: 8 WMMAs (4 col tiles x 2 K-chunks).
__device__ __forceinline__ void wmma_slab(const __bf16* sHf, const __bf16* sWf,
                                          float* __restrict__ OUT, int rowBase,
                                          int wave, int lane, int n, bool fullTile) {
  const int m = lane & 15, half = lane >> 4;
  const int rloc = wave * 16 + m;
  const bf16x16* aP = (const bf16x16*)sHf;
  const bf16x16* bP = (const bf16x16*)sWf;

  f32x8 acc0 = {}; f32x8 acc1 = {}; f32x8 acc2 = {}; f32x8 acc3 = {};
#pragma unroll
  for (int kk32 = 0; kk32 < 2; ++kk32) {
    bf16x16 af = aP[(rloc * 2 + kk32) * 2 + half];
#pragma unroll
    for (int ct = 0; ct < 4; ++ct) {
      bf16x16 bf = bP[((kk32 * 4 + ct) * 16 + m) * 2 + half];
      f32x8 c = (ct == 0) ? acc0 : (ct == 1) ? acc1 : (ct == 2) ? acc2 : acc3;
      c = __builtin_amdgcn_wmma_f32_16x16x32_bf16(false, af, false, bf,
                                                  (short)0, c, false, false);
      if (ct == 0) acc0 = c; else if (ct == 1) acc1 = c;
      else if (ct == 2) acc2 = c; else acc3 = c;
    }
  }
  // C/D layout: VGPR r holds M = r + 8*half, N = lane&15
  const int rowTop = rowBase + wave * 16 + 8 * half;
  if (fullTile) {
    // uniform fast path: no per-lane guards -> plain global_store_b32 stream
#pragma unroll
    for (int ct = 0; ct < 4; ++ct) {
      f32x8 c = (ct == 0) ? acc0 : (ct == 1) ? acc1 : (ct == 2) ? acc2 : acc3;
#pragma unroll
      for (int r = 0; r < 8; ++r)
        OUT[(size_t)(rowTop + r) * FEAT + ct * 16 + m] = c[r];
    }
  } else {
#pragma unroll
    for (int ct = 0; ct < 4; ++ct) {
      f32x8 c = (ct == 0) ? acc0 : (ct == 1) ? acc1 : (ct == 2) ? acc2 : acc3;
#pragma unroll
      for (int r = 0; r < 8; ++r) {
        int row = rowTop + r;
        if (row < n) OUT[(size_t)row * FEAT + ct * 16 + m] = c[r];
      }
    }
  }
}

// OUT = H @ W  (single weight matrix)
__global__ void __launch_bounds__(128)
gemm64_wmma(const float* __restrict__ H, const float* __restrict__ W,
            float* __restrict__ OUT, int n) {
  __shared__ __bf16 sHf[FEAT * FEAT];
  __shared__ __bf16 sWf[FEAT * FEAT];
  const int tid = threadIdx.x;
  const int rowBase = blockIdx.x * 64;
  const bool fullTile = (rowBase + 64) <= n;   // uniform (scalar) predicate
  stage_H(H, sHf, rowBase, n, tid);
  stage_W(W, sWf, tid);
  __syncthreads();
  wmma_slab(sHf, sWf, OUT, rowBase, tid >> 5, tid & 31, n, fullTile);
}

// O0 = H@W0, O1 = H@W1, O2 = H@W2 : A tile staged once, reused 3x.
__global__ void __launch_bounds__(128)
gemm64x3_wmma(const float* __restrict__ H,
              const float* __restrict__ W0, const float* __restrict__ W1,
              const float* __restrict__ W2,
              float* __restrict__ O0, float* __restrict__ O1, float* __restrict__ O2,
              int n) {
  __shared__ __bf16 sHf[FEAT * FEAT];
  __shared__ __bf16 sWf[FEAT * FEAT];
  const int tid = threadIdx.x;
  const int rowBase = blockIdx.x * 64;
  const bool fullTile = (rowBase + 64) <= n;
  stage_H(H, sHf, rowBase, n, tid);
#pragma unroll
  for (int w = 0; w < 3; ++w) {
    const float* W = (w == 0) ? W0 : (w == 1) ? W1 : W2;
    float* O       = (w == 0) ? O0 : (w == 1) ? O1 : O2;
    if (w > 0) __syncthreads();        // previous sWf readers done
    stage_W(W, sWf, tid);
    __syncthreads();                   // sWf (and on w==0, sHf) visible
    wmma_slab(sHf, sWf, O, rowBase, tid >> 5, tid & 31, n, fullTile);
  }
}

// ---------------------------------------------------------------------------
// Small / elementwise kernels
// ---------------------------------------------------------------------------
__global__ void fillz(float* __restrict__ p, long long cnt) {
  long long i = (long long)blockIdx.x * blockDim.x + threadIdx.x;
  if (i < cnt) p[i] = 0.f;
}

__global__ void softmax5(const float* __restrict__ alphas, float* __restrict__ w5, int nMix) {
  int k = threadIdx.x;
  if (k >= nMix) return;
  float a[5], mx = -1e30f;
#pragma unroll
  for (int i = 0; i < 5; ++i) { a[i] = alphas[k * 5 + i]; mx = fmaxf(mx, a[i]); }
  float s = 0.f;
#pragma unroll
  for (int i = 0; i < 5; ++i) { a[i] = __expf(a[i] - mx); s += a[i]; }
#pragma unroll
  for (int i = 0; i < 5; ++i) w5[k * 5 + i] = a[i] / s;
}

__global__ void deg_init(float* __restrict__ deg, float* __restrict__ cnt, int n) {
  int v = blockIdx.x * blockDim.x + threadIdx.x;
  if (v < n) { deg[v] = 1.f; cnt[v] = 0.f; }   // self-loop counts toward deg
}
__global__ void deg_edges(const long long* __restrict__ ei, int E,
                          float* __restrict__ deg, float* __restrict__ cnt) {
  int e = blockIdx.x * blockDim.x + threadIdx.x;
  if (e >= E) return;
  int d = (int)ei[(size_t)E + e];
  atomicAdd(&deg[d], 1.f);
  atomicAdd(&cnt[d], 1.f);
}
__global__ void deg_final(float* __restrict__ dinv, float* __restrict__ cntInv, int n) {
  int v = blockIdx.x * blockDim.x + threadIdx.x;
  if (v >= n) return;
  dinv[v]   = rsqrtf(dinv[v]);                 // deg >= 1 always (self loops)
  cntInv[v] = 1.f / fmaxf(cntInv[v], 1.f);
}

// SAGE neighbor sum: AGG[dst] += h[src]  (4 threads per edge, 16 cols each)
__global__ void sage_scatter(const long long* __restrict__ ei, int E,
                             const float* __restrict__ h, float* __restrict__ agg) {
  long long t = (long long)blockIdx.x * blockDim.x + threadIdx.x;
  int e = (int)(t >> 2);
  if (e >= E) return;
  int q = (int)(t & 3);
  int s = (int)ei[e], d = (int)ei[(size_t)E + e];
  const float* hr = h + (size_t)s * FEAT + q * 16;
  float* out = agg + (size_t)d * FEAT + q * 16;
#pragma unroll
  for (int i = 0; i < 16; ++i) atomicAdd(out + i, hr[i]);
}

__global__ void agg_scale(float* __restrict__ agg, const float* __restrict__ cntInv, int n) {
  long long t = (long long)blockIdx.x * blockDim.x + threadIdx.x;
  int v = (int)(t >> 6);
  if (v >= n) return;
  agg[t] *= cntInv[v];
}

// GAT per-node scores + emax init (self-loop score) + denom clear
__global__ void gat_scores(const float* __restrict__ HWa, const float* __restrict__ aSrc,
                           const float* __restrict__ aDst, float* __restrict__ sSrc,
                           float* __restrict__ sDst, unsigned* __restrict__ emaxEnc,
                           float* __restrict__ denom, int n) {
  int v = blockIdx.x * blockDim.x + threadIdx.x;
  if (v >= n) return;
  const float4* row = (const float4*)(HWa + (size_t)v * FEAT);
  const float4* as  = (const float4*)aSrc;
  const float4* ad  = (const float4*)aDst;
  float ss = 0.f, sd = 0.f;
#pragma unroll
  for (int i = 0; i < 16; ++i) {
    float4 h4 = row[i], a4 = as[i], d4 = ad[i];
    ss += h4.x * a4.x + h4.y * a4.y + h4.z * a4.z + h4.w * a4.w;
    sd += h4.x * d4.x + h4.y * d4.y + h4.z * d4.z + h4.w * d4.w;
  }
  sSrc[v] = ss; sDst[v] = sd;
  emaxEnc[v] = encf(leaky(ss + sd));   // self-loop edge seeds the segment max
  denom[v] = 0.f;
}

__global__ void gat_emax(const long long* __restrict__ ei, int E,
                         const float* __restrict__ sSrc, const float* __restrict__ sDst,
                         unsigned* __restrict__ emaxEnc) {
  int e = blockIdx.x * blockDim.x + threadIdx.x;
  if (e >= E) return;
  int s = (int)ei[e], d = (int)ei[(size_t)E + e];
  atomicMax(&emaxEnc[d], encf(leaky(sSrc[s] + sDst[d])));
}

__global__ void gat_denom_self(const float* __restrict__ sSrc, const float* __restrict__ sDst,
                               const unsigned* __restrict__ emaxEnc, float* __restrict__ denom,
                               int n) {
  int v = blockIdx.x * blockDim.x + threadIdx.x;
  if (v >= n) return;
  denom[v] = __expf(leaky(sSrc[v] + sDst[v]) - decf(emaxEnc[v]));
}

__global__ void gat_denom(const long long* __restrict__ ei, int E,
                          const float* __restrict__ sSrc, const float* __restrict__ sDst,
                          const unsigned* __restrict__ emaxEnc, float* __restrict__ denom) {
  int e = blockIdx.x * blockDim.x + threadIdx.x;
  if (e >= E) return;
  int s = (int)ei[e], d = (int)ei[(size_t)E + e];
  atomicAdd(&denom[d], __expf(leaky(sSrc[s] + sDst[d]) - decf(emaxEnc[d])));
}

// Fused GCN + GAT edge scatter: S[dst] += w0*norm*HWg[src] + w2*alpha*HWa[src]
__global__ void edge_scatter(const long long* __restrict__ ei, int E,
                             const float* __restrict__ HWg, const float* __restrict__ HWa,
                             const float* __restrict__ dinv, const float* __restrict__ sSrc,
                             const float* __restrict__ sDst,
                             const unsigned* __restrict__ emaxEnc,
                             const float* __restrict__ denom,
                             const float* __restrict__ w5, int k,
                             float* __restrict__ S) {
  long long t = (long long)blockIdx.x * blockDim.x + threadIdx.x;
  int e = (int)(t >> 2);
  if (e >= E) return;
  int q = (int)(t & 3);
  int s = (int)ei[e], d = (int)ei[(size_t)E + e];
  const float* rg = HWg + (size_t)s * FEAT + q * 16;
  const float* ra = HWa + (size_t)s * FEAT + q * 16;
  __builtin_prefetch(rg, 0, 0);        // global_prefetch_b8: hide gather latency
  __builtin_prefetch(ra, 0, 0);
  float cg = w5[k * 5 + 0] * dinv[s] * dinv[d];
  float x  = leaky(sSrc[s] + sDst[d]);
  float ca = w5[k * 5 + 2] * __expf(x - decf(emaxEnc[d])) / denom[d];
  float* out = S + (size_t)d * FEAT + q * 16;
#pragma unroll
  for (int i = 0; i < 16; ++i) atomicAdd(out + i, cg * rg[i] + ca * ra[i]);
}

// Self-loop terms + SAGE + identity + biases (runs after edge_scatter; plain add)
__global__ void node_final(const float* __restrict__ h, const float* __restrict__ HWg,
                           const float* __restrict__ HWa, const float* __restrict__ HR,
                           const float* __restrict__ AGGW, const float* __restrict__ dinv,
                           const float* __restrict__ sSrc, const float* __restrict__ sDst,
                           const unsigned* __restrict__ emaxEnc, const float* __restrict__ denom,
                           const float* __restrict__ gcnB, const float* __restrict__ sageB,
                           const float* __restrict__ gatB, const float* __restrict__ w5,
                           int k, float* __restrict__ S, int n) {
  long long t = (long long)blockIdx.x * blockDim.x + threadIdx.x;
  int v = (int)(t >> 6);
  if (v >= n) return;
  int c = (int)(t & 63);
  float w0 = w5[k * 5], w1 = w5[k * 5 + 1], w2 = w5[k * 5 + 2], w3 = w5[k * 5 + 3];
  float dv = dinv[v];
  float aSelf = __expf(leaky(sSrc[v] + sDst[v]) - decf(emaxEnc[v])) / denom[v];
  size_t i = (size_t)v * FEAT + c;
  S[i] += w0 * (dv * dv * HWg[i] + gcnB[c])
        + w1 * (AGGW[i] + HR[i] + sageB[c])
        + w2 * (aSelf * HWa[i] + gatB[c])
        + w3 * h[i];
}

__global__ void avg4(const float* __restrict__ a, const float* __restrict__ b,
                     const float* __restrict__ c, const float* __restrict__ d,
                     float* __restrict__ o, long long cnt) {
  long long i = (long long)blockIdx.x * blockDim.x + threadIdx.x;
  if (i < cnt) o[i] = 0.25f * (a[i] + b[i] + c[i] + d[i]);
}

// ---------------------------------------------------------------------------
// Orchestration
// ---------------------------------------------------------------------------
extern "C" void kernel_launch(void* const* d_in, const int* in_sizes, int n_in,
                              void* d_out, int out_size, void* d_ws, size_t ws_size,
                              hipStream_t stream) {
  const float*     x      = (const float*)d_in[0];
  const long long* ei     = (const long long*)d_in[1];
  const float*     alphas = (const float*)d_in[2];
  const float*     gcnW   = (const float*)d_in[3];
  const float*     gcnB   = (const float*)d_in[4];
  const float*     sageWl = (const float*)d_in[5];
  const float*     sageWr = (const float*)d_in[6];
  const float*     sageB  = (const float*)d_in[7];
  const float*     gatW   = (const float*)d_in[8];
  const float*     gatAs  = (const float*)d_in[9];
  const float*     gatAd  = (const float*)d_in[10];
  const float*     gatB   = (const float*)d_in[11];

  const int n    = in_sizes[0] / FEAT;
  const int E    = in_sizes[1] / 2;
  const int nMix = in_sizes[2] / 5;
  const long long n64 = (long long)n * FEAT;

  // workspace layout
  float* base = (float*)d_ws;
  float* st[4];
  for (int i = 0; i < 4; ++i) st[i] = base + (size_t)i * n64;
  float* HWg  = base + 4 * (size_t)n64;
  float* HWa  = base + 5 * (size_t)n64;
  float* HR   = base + 6 * (size_t)n64;
  float* AGG  = base + 7 * (size_t)n64;
  float* AGGW = base + 8 * (size_t)n64;
  float* dinv   = base + 9 * (size_t)n64;
  float* cntInv = dinv + n;
  float* sSrc   = cntInv + n;
  float* sDst   = sSrc + n;
  float* denom  = sDst + n;
  unsigned* emaxEnc = (unsigned*)(denom + n);
  float* w5 = (float*)(emaxEnc + n);

  const dim3 blk(256);
  const int gN    = (n + 255) / 256;
  const int gE    = (E + 255) / 256;
  const int gE4   = (int)(((long long)E * 4 + 255) / 256);
  const int gN64  = (int)((n64 + 255) / 256);
  const int gGemm = (n + 63) / 64;

  softmax5<<<1, 32, 0, stream>>>(alphas, w5, nMix);
  deg_init<<<gN, blk, 0, stream>>>(dinv, cntInv, n);
  deg_edges<<<gE, blk, 0, stream>>>(ei, E, dinv, cntInv);
  deg_final<<<gN, blk, 0, stream>>>(dinv, cntInv, n);

  const float* states[6];
  states[0] = x;
  states[1] = x;
  int offset = 0;
  for (int i = 0; i < 4; ++i) {
    float* S = st[i];
    states[i + 2] = S;
    fillz<<<gN64, blk, 0, stream>>>(S, n64);
    for (int j = 0; j < i + 2; ++j) {
      const int k = offset + j;
      const float* h = states[j];

      // hw_gcn / hw_gat / h@sage_Wr share the A tile -> one fused WMMA kernel
      gemm64x3_wmma<<<gGemm, 128, 0, stream>>>(h,
          gcnW + (size_t)k * 4096, gatW + (size_t)k * 4096, sageWr + (size_t)k * 4096,
          HWg, HWa, HR, n);

      fillz<<<gN64, blk, 0, stream>>>(AGG, n64);
      sage_scatter<<<gE4, blk, 0, stream>>>(ei, E, h, AGG);
      agg_scale<<<gN64, blk, 0, stream>>>(AGG, cntInv, n);
      gemm64_wmma<<<gGemm, 128, 0, stream>>>(AGG, sageWl + (size_t)k * 4096, AGGW, n);

      gat_scores<<<gN, blk, 0, stream>>>(HWa, gatAs + (size_t)k * FEAT,
                                         gatAd + (size_t)k * FEAT,
                                         sSrc, sDst, emaxEnc, denom, n);
      gat_emax<<<gE, blk, 0, stream>>>(ei, E, sSrc, sDst, emaxEnc);
      gat_denom_self<<<gN, blk, 0, stream>>>(sSrc, sDst, emaxEnc, denom, n);
      gat_denom<<<gE, blk, 0, stream>>>(ei, E, sSrc, sDst, emaxEnc, denom);

      edge_scatter<<<gE4, blk, 0, stream>>>(ei, E, HWg, HWa, dinv, sSrc, sDst,
                                            emaxEnc, denom, w5, k, S);
      node_final<<<gN64, blk, 0, stream>>>(h, HWg, HWa, HR, AGGW, dinv, sSrc, sDst,
                                           emaxEnc, denom,
                                           gcnB  + (size_t)k * FEAT,
                                           sageB + (size_t)k * FEAT,
                                           gatB  + (size_t)k * FEAT,
                                           w5, k, S, n);
    }
    offset += i + 2;
  }

  avg4<<<gN64, blk, 0, stream>>>(st[0], st[1], st[2], st[3], (float*)d_out, n64);
}
